// QuantumGAT_22711787061445
// MI455X (gfx1250) — matrix operand order repaired and checked by reading, not compiled, not run on hardware
//
#include <hip/hip_runtime.h>
#include <hip/hip_bf16.h>

// ---------------------------------------------------------------------------
// Types for CDNA5 WMMA
// ---------------------------------------------------------------------------
typedef __attribute__((ext_vector_type(16))) __bf16 v16bf;
typedef __attribute__((ext_vector_type(8)))  __bf16 v8bf;
typedef __attribute__((ext_vector_type(8)))  float  v8f;

__device__ __forceinline__ unsigned short f2bf_bits(float f) {
    unsigned u = __float_as_uint(f);
    unsigned r = u + 0x7FFFu + ((u >> 16) & 1u);   // round-to-nearest-even
    return (unsigned short)(r >> 16);
}

__device__ __forceinline__ v16bf load_v16bf(const unsigned short* p0,
                                            const unsigned short* p1) {
    v8bf lo = *(const v8bf*)p0;                    // global_load_b128
    v8bf hi = *(const v8bf*)p1;                    // global_load_b128
    return __builtin_shufflevector(lo, hi, 0,1,2,3,4,5,6,7,8,9,10,11,12,13,14,15);
}

// ordered-uint encoding so unsigned atomicMax == float max
__device__ __forceinline__ unsigned f2ord(float f) {
    unsigned u = __float_as_uint(f);
    return (u & 0x80000000u) ? ~u : (u | 0x80000000u);
}
__device__ __forceinline__ float ord2f(unsigned o) {
    unsigned u = (o & 0x80000000u) ? (o & 0x7FFFFFFFu) : ~o;
    return __uint_as_float(u);
}

// ---------------------------------------------------------------------------
// Zero fill
// ---------------------------------------------------------------------------
__global__ void zero_u32(unsigned* __restrict__ p, int n) {
    int i = blockIdx.x * blockDim.x + threadIdx.x;
    if (i < n) p[i] = 0u;
}

// ---------------------------------------------------------------------------
// Activation fp32 [N,K] -> bf16 [N,Kpad], zero padded K..Kpad
// ---------------------------------------------------------------------------
__global__ void cvt_act_bf16(const float* __restrict__ A, unsigned short* __restrict__ O,
                             int N, int K, int Kpad) {
    int idx = blockIdx.x * blockDim.x + threadIdx.x;
    if (idx >= N * Kpad) return;
    int n = idx / Kpad, k = idx - n * Kpad;
    O[idx] = (k < K) ? f2bf_bits(A[(size_t)n * K + k]) : (unsigned short)0;
}

// ---------------------------------------------------------------------------
// Pack W [K,M] fp32 (row-major) into WMMA B-fragment order (bf16):
//   Bp[((ct*KS + ks)*32 + lane)*16 + j] = W[ks*32 + (lane<16?0:16) + j][ct*16 + (lane&15)]
// so each lane's 16-element fragment is 32 contiguous bytes.
// ---------------------------------------------------------------------------
__global__ void pack_B(const float* __restrict__ W, unsigned short* __restrict__ Bp,
                       int K, int Kpad, int M) {
    int idx = blockIdx.x * blockDim.x + threadIdx.x;
    int KS = Kpad >> 5;
    int total = (M >> 4) * KS * 512;
    if (idx >= total) return;
    int j    = idx & 15;
    int lane = (idx >> 4) & 31;
    int g    = idx >> 9;
    int ks   = g % KS;
    int ct   = g / KS;
    int k    = (ks << 5) + ((lane < 16) ? 0 : 16) + j;
    int col  = (ct << 4) + (lane & 15);
    unsigned short v = 0;
    if (k < K) v = f2bf_bits(W[(size_t)k * M + col]);
    Bp[idx] = v;
}

// ---------------------------------------------------------------------------
// WMMA GEMM: C[N,M] = Abf[N,Kpad] x B (packed). Block = 128 threads (4 waves).
// Each wave: 16 rows x NT*16 cols (NT compile-time => straight-line inner
// loop: 2 A b128 loads + 2*NT B b128 loads + NT v_wmma, no exec masking).
// ---------------------------------------------------------------------------
template<int NT>
__global__ void __launch_bounds__(128)
gemm_wmma_bf16(const unsigned short* __restrict__ Abf,
               const unsigned short* __restrict__ Bp,
               float* __restrict__ C, int N, int Kpad, int M)
{
    const int wave = threadIdx.x >> 5;
    const int lane = threadIdx.x & 31;
    const int row0 = blockIdx.x << 4;
    const int colBase = (blockIdx.y * 4 + wave) * (NT * 16);
    if (colBase >= M) return;                       // wave-uniform; no LDS/sync here
    const int KS = Kpad >> 5;

    const int rowA = row0 + (lane & 15);
    const unsigned short* aptr = Abf + (size_t)rowA * Kpad + ((lane < 16) ? 0 : 16);
    const unsigned short* bptr = Bp + (size_t)(colBase >> 4) * KS * 512 + lane * 16;
    const size_t tStride = (size_t)KS * 512;        // packed elems per 16-col tile

    v8f acc[NT];
    #pragma unroll
    for (int t = 0; t < NT; ++t) acc[t] = (v8f){};

    for (int ks = 0; ks < KS; ++ks) {
        const unsigned short* ap = aptr + (ks << 5);
        v16bf a = load_v16bf(ap, ap + 8);
        const unsigned short* b0 = bptr + (size_t)ks * 512;
        #pragma unroll
        for (int t = 0; t < NT; ++t) {
            const unsigned short* bt = b0 + (size_t)t * tStride;
            v16bf b = load_v16bf(bt, bt + 8);
            acc[t] = __builtin_amdgcn_wmma_f32_16x16x32_bf16(
                false, a, false, b, (short)0, acc[t], false, false);
        }
    }

    // C/D layout: VGPR r -> row = r + (lane<16 ? 0 : 8), col = lane&15
    const int rbase = row0 + ((lane < 16) ? 0 : 8);
    #pragma unroll
    for (int t = 0; t < NT; ++t) {
        int cc = colBase + (t << 4) + (lane & 15);
        #pragma unroll
        for (int r = 0; r < 8; ++r) {
            int gr = rbase + r;
            if (gr < N) C[(size_t)gr * M + cc] = acc[t][r];
        }
    }
}

// ---------------------------------------------------------------------------
// Per-node attention scores: asn[n,h] = <h[n,h,:], a_src[h,:]>, same for adn
// ---------------------------------------------------------------------------
__global__ void scores_kernel(const float* __restrict__ h,
                              const float* __restrict__ a_src,
                              const float* __restrict__ a_dst,
                              float* __restrict__ asn, float* __restrict__ adn,
                              int n, int heads, int ch)
{
    int t = blockIdx.x * blockDim.x + threadIdx.x;
    if (t >= n * heads) return;
    int node = t / heads, hh = t - node * heads;
    const float* hp = h + (size_t)node * heads * ch + (size_t)hh * ch;
    const float* sp = a_src + hh * ch;
    const float* dp = a_dst + hh * ch;
    float s = 0.0f, d = 0.0f;
    for (int c = 0; c < ch; ++c) {
        float v = hp[c];
        s += v * sp[c];
        d += v * dp[c];
    }
    asn[t] = s; adn[t] = d;
}

// ---------------------------------------------------------------------------
// Edge pass A: leaky-relu score per (edge, head); store + atomic max per dst
// Edges e<E come from edge_index; e in [E, E+n) are self loops.
// ---------------------------------------------------------------------------
__global__ void edge_scores(const int* __restrict__ ei, int E, int n, int heads,
                            const float* __restrict__ asn, const float* __restrict__ adn,
                            float* __restrict__ sbuf, unsigned* __restrict__ mOrd)
{
    int e = blockIdx.x * blockDim.x + threadIdx.x;
    if (e >= E + n) return;
    int s = (e < E) ? ei[e]     : (e - E);
    int d = (e < E) ? ei[E + e] : (e - E);
    for (int hh = 0; hh < heads; ++hh) {
        float v = asn[s * heads + hh] + adn[d * heads + hh];
        v = (v > 0.0f) ? v : 0.2f * v;              // leaky relu
        sbuf[(size_t)e * heads + hh] = v;
        atomicMax(&mOrd[d * heads + hh], f2ord(v));
    }
}

// ---------------------------------------------------------------------------
// Edge pass B: ex = exp(s - max[dst]); accumulate denom per dst
// ---------------------------------------------------------------------------
__global__ void edge_exp(const int* __restrict__ ei, int E, int n, int heads,
                         float* __restrict__ sbuf, const unsigned* __restrict__ mOrd,
                         float* __restrict__ denom)
{
    int e = blockIdx.x * blockDim.x + threadIdx.x;
    if (e >= E + n) return;
    int d = (e < E) ? ei[E + e] : (e - E);
    for (int hh = 0; hh < heads; ++hh) {
        float v = sbuf[(size_t)e * heads + hh];
        float m = ord2f(mOrd[d * heads + hh]);
        float ex = expf(v - m);
        sbuf[(size_t)e * heads + hh] = ex;
        atomicAdd(&denom[d * heads + hh], ex);
    }
}

// ---------------------------------------------------------------------------
// Edge pass C: out[dst] += alpha * h[src]; one wave32 per edge, lanes stride
// the heads*ch channels. h-buffers are L2-resident (102MB < 192MB L2).
// ---------------------------------------------------------------------------
__global__ void edge_aggregate(const int* __restrict__ ei, int E, int n, int heads, int ch,
                               const float* __restrict__ sbuf, const float* __restrict__ denom,
                               const float* __restrict__ h, float* __restrict__ out)
{
    int wid  = (blockIdx.x * blockDim.x + threadIdx.x) >> 5;
    int lane = threadIdx.x & 31;
    if (wid >= E + n) return;
    int e = wid;
    int s = (e < E) ? ei[e]     : (e - E);
    int d = (e < E) ? ei[E + e] : (e - E);
    int HC = heads * ch;
    const float* hs = h + (size_t)s * HC;
    float* od = out + (size_t)d * HC;
    for (int c = lane; c < HC; c += 32) {
        int hh = c / ch;
        float alpha = sbuf[(size_t)e * heads + hh] /
                      (denom[d * heads + hh] + 1e-16f);
        atomicAdd(&od[c], alpha * hs[c]);
    }
}

// ---------------------------------------------------------------------------
// out = elu(out + bias), in place
// ---------------------------------------------------------------------------
__global__ void bias_elu(float* __restrict__ buf, const float* __restrict__ b,
                         int n, int HC)
{
    int i = blockIdx.x * blockDim.x + threadIdx.x;
    if (i >= n * HC) return;
    int c = i % HC;
    float v = buf[i] + b[c];
    buf[i] = (v > 0.0f) ? v : (expf(v) - 1.0f);
}

// ---------------------------------------------------------------------------
// Global mean pool (accumulate phase)
// ---------------------------------------------------------------------------
__global__ void pool_sum(const float* __restrict__ act, const int* __restrict__ batch,
                         float* __restrict__ sums, float* __restrict__ cnt,
                         int n, int ch)
{
    int t = blockIdx.x * blockDim.x + threadIdx.x;
    if (t >= n * ch) return;
    int node = t / ch, c = t - node * ch;
    int g = batch[node];
    atomicAdd(&sums[g * ch + c], act[t]);
    if (c == 0) atomicAdd(&cnt[g], 1.0f);
}

// ---------------------------------------------------------------------------
// MLP head: 32 -> 64 -> 32 -> 1, sigmoid. One thread per graph (G=256).
// ---------------------------------------------------------------------------
__global__ void mlp_head(const float* __restrict__ sums, const float* __restrict__ cnt,
                         const float* __restrict__ fw1, const float* __restrict__ fb1,
                         const float* __restrict__ fw2, const float* __restrict__ fb2,
                         const float* __restrict__ fw3, const float* __restrict__ fb3,
                         float* __restrict__ out, int G)
{
    int g = blockIdx.x * blockDim.x + threadIdx.x;
    if (g >= G) return;
    float x[32], y[64], z[32];
    float c = fmaxf(cnt[g], 1.0f);
    for (int i = 0; i < 32; ++i) x[i] = sums[g * 32 + i] / c;
    for (int j = 0; j < 64; ++j) {
        float a = fb1[j];
        for (int i = 0; i < 32; ++i) a += x[i] * fw1[i * 64 + j];
        y[j] = fmaxf(a, 0.0f);
    }
    for (int j = 0; j < 32; ++j) {
        float a = fb2[j];
        for (int i = 0; i < 64; ++i) a += y[i] * fw2[i * 32 + j];
        z[j] = fmaxf(a, 0.0f);
    }
    float a = fb3[0];
    for (int i = 0; i < 32; ++i) a += z[i] * fw3[i];
    out[g] = 1.0f / (1.0f + expf(-a));
}

// ---------------------------------------------------------------------------
// Host side
// ---------------------------------------------------------------------------
extern "C" void kernel_launch(void* const* d_in, const int* in_sizes, int n_in,
                              void* d_out, int out_size, void* d_ws, size_t ws_size,
                              hipStream_t stream)
{
    const float* x     = (const float*)d_in[0];
    const int*   ei    = (const int*)  d_in[1];
    const int*   batch = (const int*)  d_in[2];
    const float* w1  = (const float*)d_in[3];
    const float* as1 = (const float*)d_in[4];
    const float* ad1 = (const float*)d_in[5];
    const float* b1  = (const float*)d_in[6];
    const float* w2  = (const float*)d_in[7];
    const float* as2 = (const float*)d_in[8];
    const float* ad2 = (const float*)d_in[9];
    const float* b2  = (const float*)d_in[10];
    const float* w3  = (const float*)d_in[11];
    const float* as3 = (const float*)d_in[12];
    const float* ad3 = (const float*)d_in[13];
    const float* b3  = (const float*)d_in[14];
    const float* fw1 = (const float*)d_in[15];
    const float* fb1 = (const float*)d_in[16];
    const float* fw2 = (const float*)d_in[17];
    const float* fb2 = (const float*)d_in[18];
    const float* fw3 = (const float*)d_in[19];
    const float* fb3 = (const float*)d_in[20];

    const int N_ = in_sizes[0] / 21;     // 100000
    const int E_ = in_sizes[1] / 2;      // 1600000
    const int G_ = out_size;             // 256
    const int ET = E_ + N_;              // edges + self loops

    // workspace carving (256B aligned)
    size_t off = 0;
    auto carve = [&](size_t bytes) -> void* {
        void* p = (char*)d_ws + off;
        off += (bytes + 255) & ~(size_t)255;
        return p;
    };
    float*          P     = (float*)carve((size_t)N_ * 256 * 4); // transform buffer
    float*          Q     = (float*)carve((size_t)N_ * 256 * 4); // act / aggregate buffer
    unsigned short* Abf   = (unsigned short*)carve((size_t)N_ * 256 * 2); // bf16 acts
    float*          asn   = (float*)carve((size_t)N_ * 4 * 4);
    float*          adn   = (float*)carve((size_t)N_ * 4 * 4);
    unsigned*       mOrd  = (unsigned*)carve((size_t)N_ * 4 * 4);
    float*          denom = (float*)carve((size_t)N_ * 4 * 4);
    float*          sbuf  = (float*)carve((size_t)ET * 4 * 4);
    unsigned short* Bp1   = (unsigned short*)carve((size_t)32  * 256 * 2);
    unsigned short* Bp2   = (unsigned short*)carve((size_t)256 * 256 * 2);
    unsigned short* Bp3   = (unsigned short*)carve((size_t)256 * 32  * 2);
    float*          sums  = (float*)carve((size_t)G_ * 32 * 4);
    float*          cnt   = (float*)carve((size_t)G_ * 4);
    (void)ws_size; (void)n_in;

    // pre-pack weights into WMMA B-fragment order (bf16)
    pack_B<<<(32 * 256 + 255) / 256, 256, 0, stream>>>(w1, Bp1, 21, 32, 256);
    pack_B<<<(256 * 256 + 255) / 256, 256, 0, stream>>>(w2, Bp2, 256, 256, 256);
    pack_B<<<(256 * 32 + 255) / 256, 256, 0, stream>>>(w3, Bp3, 256, 256, 32);

    auto run_layer = [&](const float* act_in, int Kin, int Kpad,
                         const unsigned short* Bp, int Cout, int heads, int ch,
                         const float* a_s, const float* a_d, const float* bias,
                         float* hbuf, float* outbuf) {
        // 0) activations -> bf16 (zero-padded to Kpad)
        int na = N_ * Kpad;
        cvt_act_bf16<<<(na + 255) / 256, 256, 0, stream>>>(act_in, Abf, N_, Kin, Kpad);
        // 1) h = act_in @ W   (WMMA); NT chosen at compile time per output width
        if (Cout == 256) {
            dim3 gg((N_ + 15) / 16, 1);                 // 4 waves x 64 cols = 256
            gemm_wmma_bf16<4><<<gg, 128, 0, stream>>>(Abf, Bp, hbuf, N_, Kpad, Cout);
        } else {
            dim3 gg((N_ + 15) / 16, 1);                 // waves 0,1 cover 32 cols
            gemm_wmma_bf16<1><<<gg, 128, 0, stream>>>(Abf, Bp, hbuf, N_, Kpad, Cout);
        }
        // 2) per-node scores
        int nh = N_ * heads;
        scores_kernel<<<(nh + 255) / 256, 256, 0, stream>>>(hbuf, a_s, a_d, asn, adn,
                                                            N_, heads, ch);
        // 3) zero accumulators (aggregate buffer may alias act_in: gemm done reading)
        zero_u32<<<(nh + 255) / 256, 256, 0, stream>>>(mOrd, nh);
        zero_u32<<<(nh + 255) / 256, 256, 0, stream>>>((unsigned*)denom, nh);
        int nc = N_ * Cout;
        zero_u32<<<(nc + 255) / 256, 256, 0, stream>>>((unsigned*)outbuf, nc);
        // 4) segment max, 5) exp + denom, 6) weighted scatter
        edge_scores<<<(ET + 255) / 256, 256, 0, stream>>>(ei, E_, N_, heads,
                                                          asn, adn, sbuf, mOrd);
        edge_exp<<<(ET + 255) / 256, 256, 0, stream>>>(ei, E_, N_, heads,
                                                       sbuf, mOrd, denom);
        int aggBlocks = (ET + 7) / 8;    // 8 waves of 32 per 256-thread block
        edge_aggregate<<<aggBlocks, 256, 0, stream>>>(ei, E_, N_, heads, ch,
                                                      sbuf, denom, hbuf, outbuf);
        // 7) bias + elu
        bias_elu<<<(nc + 255) / 256, 256, 0, stream>>>(outbuf, bias, N_, Cout);
    };

    run_layer(x, 21,  32,  Bp1, 256, 4, 64, as1, ad1, b1, P, Q);   // act1 = Q
    run_layer(Q, 256, 256, Bp2, 256, 4, 64, as2, ad2, b2, P, Q);   // act2 = Q
    run_layer(Q, 256, 256, Bp3, 32,  1, 32, as3, ad3, b3, P, Q);   // act3 = Q (N x 32)

    // pooling + head
    zero_u32<<<(G_ * 32 + 255) / 256, 256, 0, stream>>>((unsigned*)sums, G_ * 32);
    zero_u32<<<(G_ + 255) / 256, 256, 0, stream>>>((unsigned*)cnt, G_);
    int pc = N_ * 32;
    pool_sum<<<(pc + 255) / 256, 256, 0, stream>>>(Q, batch, sums, cnt, N_, 32);
    mlp_head<<<(G_ + 63) / 64, 64, 0, stream>>>(sums, cnt, fw1, fb1, fw2, fb2,
                                                fw3, fb3, (float*)d_out, G_);
}
